// CvaeEncoder_24713241821963
// MI455X (gfx1250) — matrix operand order repaired and checked by reading, not compiled, not run on hardware
//
#include <hip/hip_runtime.h>
#include <hip/hip_bf16.h>
#include <cstdint>

typedef _Float16 f16;
typedef __attribute__((ext_vector_type(16))) _Float16 v16h;
typedef __attribute__((ext_vector_type(8)))  _Float16 f16x8;
typedef __attribute__((ext_vector_type(8)))  float    v8f;
typedef __attribute__((ext_vector_type(4)))  int      i32x4;

#define BATCH 8
#define NPTS  4096
#define KNN   20
#define EMB   1024

#define PTS_PER_WG 4
#define EROWS (PTS_PER_WG * KNN)   // 80 rows = 5 M-tiles of 16

// ---- CDNA5 async global->LDS path (ASYNCcnt), with safe fallback -----------
#if defined(__has_builtin)
# if __has_builtin(__builtin_amdgcn_global_load_async_to_lds_b128) && \
     __has_builtin(__builtin_amdgcn_s_wait_asynccnt)
#  define HAVE_ASYNC_LDS 1
# endif
#endif
#ifndef HAVE_ASYNC_LDS
# define HAVE_ASYNC_LDS 0
#endif

__device__ __forceinline__ void copy16_to_lds(void* lds, const void* g) {
#if HAVE_ASYNC_LDS
  // signature (from hipcc diagnostic): (int4 AS1* gsrc, int4 AS3* ldsdst, imm, imm)
  __builtin_amdgcn_global_load_async_to_lds_b128(
      (__attribute__((address_space(1))) i32x4*)g,
      (__attribute__((address_space(3))) i32x4*)lds, 0, 0);
#else
  *(f16x8*)lds = *(const f16x8*)g;
#endif
}

__device__ __forceinline__ void async_wait_all() {
#if HAVE_ASYNC_LDS
  __builtin_amdgcn_s_wait_asynccnt(0);
#endif
}

__device__ __forceinline__ float lrelu(float v) { return v > 0.0f ? v : 0.2f * v; }
__device__ __forceinline__ float bninv() { return rsqrtf(1.0f + 1e-5f); }
__device__ __forceinline__ int wave_id() {
  return (int)(__builtin_amdgcn_readfirstlane(threadIdx.x) >> 5);   // SGPR wave index
}

// ---------------- WMMA operand loaders (v_wmma_f32_16x16x32_f16, wave32) ----
// A: 16x32 f16, row-major in LDS with stride lda. Lane l<16 holds row l,
// K = {0..7, 16..23}; lane l>=16 holds row l-16, K = {8..15, 24..31}.
__device__ __forceinline__ v16h load_a16(const f16* A, int lda, int m0, int k0) {
  const int lane = threadIdx.x & 31;
  const f16* p = A + (m0 + (lane & 15)) * lda + k0 + ((lane >> 4) << 3);
  f16x8 lo = *(const f16x8*)p;          // K offsets +0..7
  f16x8 hi = *(const f16x8*)(p + 16);   // K offsets +16..23
  v16h r;
#pragma unroll
  for (int e = 0; e < 8; ++e) { r[e] = lo[e]; r[e + 8] = hi[e]; }
  return r;
}

// B: 32x16 f16, weight stored transposed Wt[N][K] (K contiguous): lane l holds
// column l&15, K = {0..7, 16..23} (l<16) or {8..15, 24..31} (l>=16), same as A.
__device__ __forceinline__ v16h load_b16(const f16* Wt, int ldb, int n0, int k0) {
  const int lane = threadIdx.x & 31;
  const f16* p = Wt + (n0 + (lane & 15)) * ldb + k0 + ((lane >> 4) << 3);
  f16x8 lo = *(const f16x8*)p;
  f16x8 hi = *(const f16x8*)(p + 16);
  v16h r;
#pragma unroll
  for (int e = 0; e < 8; ++e) { r[e] = lo[e]; r[e + 8] = hi[e]; }
  return r;
}

// ---------------- kNN: brute force, top-20 smallest squared distance --------
__global__ void knn_kernel(const float* __restrict__ x, int* __restrict__ idx) {
  __shared__ float tile[256 * 3];
  const int b = blockIdx.y;
  const int i = blockIdx.x * 256 + threadIdx.x;     // point within batch
  const float* xb = x + (size_t)b * NPTS * 3;
  const float px = xb[i * 3 + 0], py = xb[i * 3 + 1], pz = xb[i * 3 + 2];
  float best[KNN];
  int   bidx[KNN];
#pragma unroll
  for (int s = 0; s < KNN; ++s) { best[s] = 3.0e38f; bidx[s] = 0; }
  for (int jt = 0; jt < NPTS; jt += 256) {
    __syncthreads();
    const int t = threadIdx.x;
    tile[t * 3 + 0] = xb[(jt + t) * 3 + 0];
    tile[t * 3 + 1] = xb[(jt + t) * 3 + 1];
    tile[t * 3 + 2] = xb[(jt + t) * 3 + 2];
    __syncthreads();
    for (int j = 0; j < 256; ++j) {
      float dx = px - tile[j * 3 + 0];
      float dy = py - tile[j * 3 + 1];
      float dz = pz - tile[j * 3 + 2];
      float d = dx * dx + dy * dy + dz * dz;
      if (d < best[KNN - 1]) {                       // branchless sorted insert
        const int jg = jt + j;
#pragma unroll
        for (int s = KNN - 1; s >= 1; --s) {
          bool c  = d < best[s];
          bool cp = d < best[s - 1];
          float nb = cp ? best[s - 1] : d;
          int   ni = cp ? bidx[s - 1] : jg;
          if (c) { best[s] = nb; bidx[s] = ni; }
        }
        if (d < best[0]) { best[0] = d; bidx[0] = jg; }
      }
    }
  }
  int* op = idx + ((size_t)b * NPTS + i) * KNN;
#pragma unroll
  for (int s = 0; s < KNN; ++s) op[s] = bidx[s];
}

// ---------------- weight convert + transpose: f32[cin,cout] -> f16[cout][cin]
__global__ void wcvt_kernel(const float* __restrict__ src, f16* __restrict__ dst,
                            int cin, int cout) {
  int i = blockIdx.x * 256 + threadIdx.x;
  if (i < cin * cout) {
    int c = i / cout, o = i - c * cout;
    dst[o * cin + c] = (f16)src[i];
  }
}

// ---------------- fused edge-conv: gather + L1..L4 + K-max pooling ----------
__device__ __forceinline__ void pool_to_hc(const f16* buf, int C, int colOff,
                                           f16* __restrict__ hc, int n0) {
  for (int i = threadIdx.x; i < PTS_PER_WG * C; i += 256) {
    int p = i / C, c = i - p * C;
    float m = -3.0e38f;
#pragma unroll 4
    for (int k = 0; k < KNN; ++k) m = fmaxf(m, (float)buf[(p * KNN + k) * C + c]);
    hc[(size_t)(n0 + p) * 512 + colOff + c] = (f16)m;
  }
}

__device__ __forceinline__ void layer_wmma(const f16* in, int Cin, f16* out, int Cout,
                                           const f16* Wt,
                                           const float* __restrict__ gam,
                                           const float* __restrict__ bet) {
  const int wave = wave_id();                        // scalar -> EXEC stays full
  const int lane = threadIdx.x & 31;
  const int ntiles = Cout >> 4;
  const int total  = (EROWS / 16) * ntiles;
  const float inv = bninv();
  for (int t = wave; t < total; t += 8) {
    const int mt = t / ntiles, nt = t - mt * ntiles;
    v8f acc = {};
    for (int k0 = 0; k0 < Cin; k0 += 32) {
      v16h a = load_a16(in, Cin, mt * 16, k0);
      v16h b = load_b16(Wt, Cin, nt * 16, k0);
      acc = __builtin_amdgcn_wmma_f32_16x16x32_f16(false, a, false, b,
                                                   (short)0, acc, false, false);
    }
    const int n = nt * 16 + (lane & 15);
    const float s = gam[n] * inv, bb = bet[n];
    const int mb = mt * 16 + ((lane >> 4) << 3);   // C/D layout: vgpr r -> M = mb+r
#pragma unroll
    for (int r = 0; r < 8; ++r)
      out[(mb + r) * Cout + n] = (f16)lrelu(acc[r] * s + bb);
  }
}

__global__ void edge_kernel(const float* __restrict__ x, const int* __restrict__ idx,
                            const float* __restrict__ W1, const float* __restrict__ g1, const float* __restrict__ b1,
                            const f16* __restrict__ wAll,   // W2t|W3t|W4t contiguous f16
                            const float* __restrict__ g2, const float* __restrict__ b2,
                            const float* __restrict__ g3, const float* __restrict__ b3,
                            const float* __restrict__ g4, const float* __restrict__ b4,
                            f16* __restrict__ hc) {
  extern __shared__ char smem[];
  f16* bufA  = (f16*)smem;                  // EROWS*128
  f16* bufB  = bufA + EROWS * 128;          // EROWS*256
  f16* sW2   = bufB + EROWS * 256;          // 64*64
  f16* sW3   = sW2 + 64 * 64;               // 128*64
  f16* sW4   = sW3 + 128 * 64;              // 256*128
  float* in6 = (float*)(sW4 + 256 * 128);   // EROWS*6
  const int tid = threadIdx.x;
  const int n0  = blockIdx.x * PTS_PER_WG;  // flattened b*N + n

  // weights -> LDS (async b128 stream; overlaps with the gather below)
  {
    const int WCHUNKS = (64 * 64 + 128 * 64 + 256 * 128) / 8;  // 5632
    for (int i = tid; i < WCHUNKS; i += 256)
      copy16_to_lds(sW2 + i * 8, wAll + i * 8);
  }
  // gather [feature(neighbor), center] -> in6
  if (tid < EROWS) {
    const int p = tid / KNN, k = tid - p * KNN;
    const int nc = n0 + p;
    const int b  = nc >> 12;                 // N = 4096
    const int j  = idx[(size_t)nc * KNN + k];
    const int jg = (b << 12) + j;
    in6[tid * 6 + 0] = x[(size_t)jg * 3 + 0];
    in6[tid * 6 + 1] = x[(size_t)jg * 3 + 1];
    in6[tid * 6 + 2] = x[(size_t)jg * 3 + 2];
    in6[tid * 6 + 3] = x[(size_t)nc * 3 + 0];
    in6[tid * 6 + 4] = x[(size_t)nc * 3 + 1];
    in6[tid * 6 + 5] = x[(size_t)nc * 3 + 2];
  }
  async_wait_all();
  __syncthreads();
  // layer1: 6 -> 64, scalar (K too small for WMMA)
  const float inv = bninv();
  for (int i = tid; i < EROWS * 64; i += 256) {
    const int r = i >> 6, o = i & 63;
    float a = 0.0f;
#pragma unroll
    for (int c = 0; c < 6; ++c) a += in6[r * 6 + c] * W1[c * 64 + o];
    bufA[r * 64 + o] = (f16)lrelu(a * (g1[o] * inv) + b1[o]);
  }
  __syncthreads();
  pool_to_hc(bufA, 64, 0, hc, n0);                 // x1
  layer_wmma(bufA, 64, bufB, 64, sW2, g2, b2);     // h2
  __syncthreads();
  pool_to_hc(bufB, 64, 64, hc, n0);                // x2
  layer_wmma(bufB, 64, bufA, 128, sW3, g3, b3);    // h3
  __syncthreads();
  pool_to_hc(bufA, 128, 128, hc, n0);              // x3
  layer_wmma(bufA, 128, bufB, 256, sW4, g4, b4);   // h4
  __syncthreads();
  pool_to_hc(bufB, 256, 256, hc, n0);              // x4
}

// ---------------- layer5 GEMM: [32768,512] x [512,1024], BN + lrelu --------
__global__ void gemm_l5_kernel(const f16* __restrict__ A, const f16* __restrict__ Wt,
                               const float* __restrict__ gam, const float* __restrict__ bet,
                               f16* __restrict__ O) {
  __shared__ f16 As[128 * 32];
  __shared__ f16 Bs[64 * 32];
  const int m0 = blockIdx.x * 128;
  const int n0 = blockIdx.y * 64;
  const int tid = threadIdx.x;
  const int wave = wave_id();
  v8f acc[4];
#pragma unroll
  for (int i = 0; i < 4; ++i) { v8f z = {}; acc[i] = z; }
  for (int k0 = 0; k0 < 512; k0 += 32) {
    __syncthreads();                               // LDS reuse barrier
    for (int c = tid; c < 512; c += 256) {         // As: 128 rows x 32 K
      int r = c >> 2, kk = (c & 3) * 8;
      copy16_to_lds(As + r * 32 + kk, A + (size_t)(m0 + r) * 512 + k0 + kk);
    }
    for (int c = tid; c < 256; c += 256) {         // Bs: 64 cols x 32 K (Wt layout)
      int r = c >> 2, kk = (c & 3) * 8;
      copy16_to_lds(Bs + r * 32 + kk, Wt + (size_t)(n0 + r) * 512 + k0 + kk);
    }
    async_wait_all();
    __syncthreads();
    v16h a = load_a16(As, 32, wave * 16, 0);
#pragma unroll
    for (int nt = 0; nt < 4; ++nt) {
      v16h b = load_b16(Bs, 32, nt * 16, 0);
      acc[nt] = __builtin_amdgcn_wmma_f32_16x16x32_f16(false, a, false, b,
                                                       (short)0, acc[nt], false, false);
    }
  }
  const int lane = tid & 31;
  const int nl = lane & 15;
  const int mh = (lane >> 4) << 3;
  const float inv = bninv();
#pragma unroll
  for (int nt = 0; nt < 4; ++nt) {
    const int n = n0 + nt * 16 + nl;
    const float s = gam[n] * inv, bb = bet[n];
#pragma unroll
    for (int r = 0; r < 8; ++r) {
      const int m = m0 + wave * 16 + mh + r;
      O[(size_t)m * EMB + n] = (f16)lrelu(acc[nt][r] * s + bb);
    }
  }
}

// ---------------- global max + mean over N per (batch, 8 channels) ----------
__global__ void reduce_kernel(const f16* __restrict__ h5, float* __restrict__ g) {
  const int b = blockIdx.y;
  const int c0 = threadIdx.x * 8;                  // blockDim.x = 128 -> 1024 ch
  const f16* p = h5 + (size_t)b * NPTS * EMB + c0;
  float mx[8], sm[8];
#pragma unroll
  for (int e = 0; e < 8; ++e) { mx[e] = -3.0e38f; sm[e] = 0.0f; }
  for (int n = 0; n < NPTS; ++n) {
    f16x8 v = *(const f16x8*)(p + (size_t)n * EMB);
#pragma unroll
    for (int e = 0; e < 8; ++e) {
      float f = (float)v[e];
      mx[e] = fmaxf(mx[e], f);
      sm[e] += f;
    }
  }
#pragma unroll
  for (int e = 0; e < 8; ++e) {
    g[b * 2 * EMB + c0 + e]       = mx[e];                 // gmax
    g[b * 2 * EMB + EMB + c0 + e] = sm[e] * (1.0f / NPTS); // gavg
  }
}

// ---------------- tiny FC head ----------------------------------------------
__global__ void head_kernel(const float* __restrict__ g,
                            const float* __restrict__ L1, const float* __restrict__ g6, const float* __restrict__ b6,
                            const float* __restrict__ L2, const float* __restrict__ L2b,
                            const float* __restrict__ g7, const float* __restrict__ b7,
                            const float* __restrict__ L3, const float* __restrict__ L3b,
                            float* __restrict__ out) {
  __shared__ float sg[2048];
  __shared__ float sf1[512];
  __shared__ float sf2[256];
  const int b = blockIdx.x, tid = threadIdx.x;
  for (int i = tid; i < 2048; i += 256) sg[i] = g[b * 2048 + i];
  __syncthreads();
  const float inv = bninv();
  for (int o = tid; o < 512; o += 256) {
    float a = 0.0f;
    for (int c = 0; c < 2048; ++c) a += sg[c] * L1[c * 512 + o];
    sf1[o] = lrelu(a * (g6[o] * inv) + b6[o]);
  }
  __syncthreads();
  for (int o = tid; o < 256; o += 256) {
    float a = 0.0f;
    for (int c = 0; c < 512; ++c) a += sf1[c] * L2[c * 256 + o];
    a += L2b[o];
    sf2[o] = lrelu(a * (g7[o] * inv) + b7[o]);
  }
  __syncthreads();
  if (tid < 128) {
    float a = 0.0f;
    for (int c = 0; c < 256; ++c) a += sf2[c] * L3[c * 128 + tid];
    out[b * 128 + tid] = a + L3b[tid];
  }
}

// ---------------- launcher --------------------------------------------------
extern "C" void kernel_launch(void* const* d_in, const int* in_sizes, int n_in,
                              void* d_out, int out_size, void* d_ws, size_t ws_size,
                              hipStream_t stream) {
  const float* x   = (const float*)d_in[0];
  const float* W1  = (const float*)d_in[1];
  const float* g1  = (const float*)d_in[2];
  const float* b1  = (const float*)d_in[3];
  const float* W2  = (const float*)d_in[4];
  const float* g2  = (const float*)d_in[5];
  const float* b2  = (const float*)d_in[6];
  const float* W3  = (const float*)d_in[7];
  const float* g3  = (const float*)d_in[8];
  const float* b3  = (const float*)d_in[9];
  const float* W4  = (const float*)d_in[10];
  const float* g4  = (const float*)d_in[11];
  const float* b4  = (const float*)d_in[12];
  const float* W5  = (const float*)d_in[13];
  const float* g5  = (const float*)d_in[14];
  const float* b5  = (const float*)d_in[15];
  const float* L1  = (const float*)d_in[16];
  const float* g6  = (const float*)d_in[17];
  const float* b6  = (const float*)d_in[18];
  const float* L2  = (const float*)d_in[19];
  const float* L2b = (const float*)d_in[20];
  const float* g7  = (const float*)d_in[21];
  const float* b7  = (const float*)d_in[22];
  const float* L3  = (const float*)d_in[23];
  const float* L3b = (const float*)d_in[24];
  float* out = (float*)d_out;
  (void)in_sizes; (void)n_in; (void)out_size; (void)ws_size;

  char* ws = (char*)d_ws;
  size_t off = 0;
  auto carve = [&](size_t bytes) {
    char* p = ws + off;
    off = (off + bytes + 255) & ~(size_t)255;
    return p;
  };
  int* idxBuf = (int*)carve((size_t)BATCH * NPTS * KNN * 4);      // 2.6 MB
  f16* hcBuf  = (f16*)carve((size_t)BATCH * NPTS * 512 * 2);      // 33.5 MB
  f16* h5Buf  = (f16*)carve((size_t)BATCH * NPTS * EMB * 2);      // 67 MB
  // W2t|W3t|W4t contiguous so the edge kernel can stream one b128 sequence
  f16* wEdge  = (f16*)carve((size_t)(64 * 64 + 128 * 64 + 256 * 128) * 2);
  f16* w2t = wEdge;
  f16* w3t = wEdge + 64 * 64;
  f16* w4t = wEdge + 64 * 64 + 128 * 64;
  f16* w5t    = (f16*)carve((size_t)EMB * 512 * 2);               // 1 MB
  float* gBuf = (float*)carve((size_t)BATCH * 2 * EMB * 4);

  knn_kernel<<<dim3(NPTS / 256, BATCH), 256, 0, stream>>>(x, idxBuf);
  wcvt_kernel<<<(64 * 64 + 255) / 256, 256, 0, stream>>>(W2, w2t, 64, 64);
  wcvt_kernel<<<(64 * 128 + 255) / 256, 256, 0, stream>>>(W3, w3t, 64, 128);
  wcvt_kernel<<<(128 * 256 + 255) / 256, 256, 0, stream>>>(W4, w4t, 128, 256);
  wcvt_kernel<<<(512 * 1024 + 255) / 256, 256, 0, stream>>>(W5, w5t, 512, 1024);

  const size_t smem_edge =
      (size_t)(EROWS * 128 + EROWS * 256 + 64 * 64 + 128 * 64 + 256 * 128) * sizeof(f16)
      + (size_t)EROWS * 6 * sizeof(float);   // ~150 KB, within 320 KB/WGP
  edge_kernel<<<BATCH * NPTS / PTS_PER_WG, 256, smem_edge, stream>>>(
      x, idxBuf, W1, g1, b1, wEdge, g2, b2, g3, b3, g4, b4, hcBuf);

  gemm_l5_kernel<<<dim3(BATCH * NPTS / 128, EMB / 64), 256, 0, stream>>>(
      hcBuf, w5t, g5, b5, h5Buf);
  reduce_kernel<<<dim3(1, BATCH), 128, 0, stream>>>(h5Buf, gBuf);
  head_kernel<<<BATCH, 256, 0, stream>>>(gBuf, L1, g6, b6, L2, L2b, g7, b7, L3, L3b, out);
}
